// MultiheadAttention_23287312678913
// MI455X (gfx1250) — matrix operand order repaired and checked by reading, not compiled, not run on hardware
//
#include <hip/hip_runtime.h>
#include <hip/hip_bf16.h>

// MI455X / gfx1250, wave32. All matmuls via V_WMMA_F32_16X16X32_F16.
// LDS staging via GLOBAL_LOAD_ASYNC_TO_LDS_B128 (ASYNCcnt) when available.

typedef __attribute__((ext_vector_type(16))) _Float16 v16h;
typedef __attribute__((ext_vector_type(8)))  float    v8f;

union Frag16 { v16h v; uint4 q[2]; };
union FragC  { v8f  v; float f[8]; };

#if defined(__has_builtin)
#if __has_builtin(__builtin_amdgcn_global_load_async_to_lds_b128)
#define USE_ASYNC_LDS 1
#endif
#endif

// Builtin parameter types (from hipcc diagnostic):
//   param0: int __attribute__((vector_size(16))) addrspace(1)*  (global src)
//   param1: int __attribute__((vector_size(16))) addrspace(3)*  (LDS dst)
typedef int v4i_gcc __attribute__((vector_size(16)));
typedef __attribute__((address_space(1))) v4i_gcc as1_v4i;
typedef __attribute__((address_space(3))) v4i_gcc as3_v4i;

// Copy 16 bytes global -> LDS. Async path writes LDS directly from the memory
// hierarchy (no VGPR round-trip), tracked by ASYNCcnt.
__device__ __forceinline__ void g2lds_b128(const _Float16* g, _Float16* l) {
#if USE_ASYNC_LDS
  // generic LDS pointer low 32 bits == LDS byte offset (flat LDS aperture)
  __builtin_amdgcn_global_load_async_to_lds_b128(
      (as1_v4i*)(uintptr_t)g,
      (as3_v4i*)(unsigned int)(uintptr_t)l,
      0, 0);
#else
  *reinterpret_cast<uint4*>(l) = *reinterpret_cast<const uint4*>(g);
#endif
}

__device__ __forceinline__ void wait_async_all() {
#if USE_ASYNC_LDS
#if __has_builtin(__builtin_amdgcn_s_wait_asynccnt)
  __builtin_amdgcn_s_wait_asynccnt(0);
#else
  asm volatile("s_wait_asynccnt 0x0" ::: "memory");
#endif
#endif
}

__device__ __forceinline__ v8f wmma16(v16h a, v16h b, v8f c) {
  // 8 args: (neg_a, A, neg_b, B, c_mod, C, reuse_a, reuse_b)
  return __builtin_amdgcn_wmma_f32_16x16x32_f16(false, a, false, b, (short)0, c,
                                                false, false);
}

// A-matrix 16x32 f16 fragment from row-major [.., ld] matrix.
// Lane l: row = row_base + (l&15); halves 0..7 -> K = (l>>4)*8 + 0..7,
// halves 8..15 -> K = 16 + (l>>4)*8 + 0..7   (ISA 7.12.2 16-bit A layout)
__device__ __forceinline__ v16h load_a_frag(const _Float16* base, int ld,
                                            int row_base, int k_base) {
  const int lane = threadIdx.x & 31;
  const int m  = lane & 15;
  const int hs = lane >> 4;
  const _Float16* p = base + (size_t)(row_base + m) * ld + k_base + hs * 8;
  Frag16 u;
  u.q[0] = *reinterpret_cast<const uint4*>(p);
  u.q[1] = *reinterpret_cast<const uint4*>(p + 16);
  return u.v;
}

// B-matrix 32x16 f16 fragment, where memory holds B^T row-major:
// element (K=k, N=n) at bt[n*ld + k]. Lane l: n = n_base + (l&15);
// halves 0..7 -> K = k_base + (l>>4)*8 + 0..7, halves 8..15 -> K = k_base + 16 + (l>>4)*8 + 0..7
__device__ __forceinline__ v16h load_bT_frag(const _Float16* bt, int ld,
                                             int n_base, int k_base) {
  const int lane = threadIdx.x & 31;
  const int n  = n_base + (lane & 15);
  const int hs = lane >> 4;
  const _Float16* p = bt + (size_t)n * ld + k_base + hs * 8;
  Frag16 u;
  u.q[0] = *reinterpret_cast<const uint4*>(p);
  u.q[1] = *reinterpret_cast<const uint4*>(p + 16);
  return u.v;
}

// ---------------- conversion kernels ----------------

__global__ void f32_to_f16_k(const float* __restrict__ src,
                             _Float16* __restrict__ dst, int n) {
  int i = blockIdx.x * blockDim.x + threadIdx.x;
  if (i < n) dst[i] = (_Float16)src[i];
}

// wt[n*768 + k] = (f16) w[k*768 + n]
__global__ void wT_f32_to_f16_k(const float* __restrict__ w,
                                _Float16* __restrict__ wt) {
  int i = blockIdx.x * blockDim.x + threadIdx.x;  // i = n*768 + k
  int n = i / 768;
  int k = i - n * 768;
  wt[i] = (_Float16)w[k * 768 + n];
}

// ---------------- WMMA GEMM: C[M x 768] = A[M x 768] * B, Bt = B^T ----------
// Block: 256 thr = 8 waves. Wave w computes rows [bx*128 + w*16, +16), cols
// [by*64, +64) as 4 accumulators. Bt tile (64 n x 32 k) staged in LDS.
// MODE 0: f16 row-major [M][768]; MODE 1: f16 transposed [768][M];
// MODE 2: f32 row-major + bias.

template <int MODE>
__global__ void gemm768_wmma(const _Float16* __restrict__ A,
                             const _Float16* __restrict__ Bt,
                             _Float16* __restrict__ Of16,
                             float* __restrict__ Of32,
                             const float* __restrict__ bias, int Mrows) {
  __shared__ __align__(16) _Float16 lds_bt[64 * 32];
  const int wave = threadIdx.x >> 5;
  const int lane = threadIdx.x & 31;
  const int row_base = blockIdx.x * 128 + wave * 16;
  const int n0 = blockIdx.y * 64;

  FragC c[4];
#pragma unroll
  for (int t = 0; t < 4; ++t)
#pragma unroll
    for (int r = 0; r < 8; ++r) c[t].f[r] = 0.0f;

  const int bn = threadIdx.x >> 2;          // 0..63
  const int bp = (threadIdx.x & 3) * 8;     // 0,8,16,24

  for (int k0 = 0; k0 < 768; k0 += 32) {
    // stage Bt tile: 64 rows x 32 halves (async global -> LDS)
    g2lds_b128(&Bt[(size_t)(n0 + bn) * 768 + k0 + bp], &lds_bt[bn * 32 + bp]);
    if (k0 + 32 < 768)
      __builtin_prefetch(&Bt[(size_t)(n0 + bn) * 768 + k0 + 32 + bp], 0, 1);
    wait_async_all();
    __syncthreads();

    v16h a = load_a_frag(A, 768, row_base, k0);
    c[0].v = wmma16(a, load_bT_frag(lds_bt, 32, 0, 0), c[0].v);
    c[1].v = wmma16(a, load_bT_frag(lds_bt, 32, 16, 0), c[1].v);
    c[2].v = wmma16(a, load_bT_frag(lds_bt, 32, 32, 0), c[2].v);
    c[3].v = wmma16(a, load_bT_frag(lds_bt, 32, 48, 0), c[3].v);
    __syncthreads();
  }

  // C layout: VGPR r, lanes 0-15 -> M = r; lanes 16-31 -> M = r + 8; N = lane&15
  const int nn = lane & 15;
  const int hs = lane >> 4;
#pragma unroll
  for (int r = 0; r < 8; ++r) {
    const int m = row_base + r + hs * 8;
#pragma unroll
    for (int t = 0; t < 4; ++t) {
      const int col = n0 + t * 16 + nn;
      if constexpr (MODE == 0) {
        Of16[(size_t)m * 768 + col] = (_Float16)c[t].f[r];
      } else if constexpr (MODE == 1) {
        Of16[(size_t)col * Mrows + m] = (_Float16)c[t].f[r];
      } else {
        Of32[(size_t)m * 768 + col] = c[t].f[r] + bias[col];
      }
    }
  }
}

// ---------------- flash attention (causal, online softmax) -----------------
// Grid: (S/64, H). Block: 128 thr = 4 waves; wave w owns query rows
// qrow = bx*64 + w*16 .. +16. Loops 32-key blocks; K/V tiles staged in LDS
// cooperatively (uniform barrier count across waves), compute guarded by a
// wave-uniform causal bound so EXEC is all-1s at every WMMA.

__global__ void attn_flash_wmma(const _Float16* __restrict__ qbuf,  // [S][768]
                                const _Float16* __restrict__ kbuf,  // [S][768]
                                const _Float16* __restrict__ vt,    // [768][S]
                                _Float16* __restrict__ attb) {      // [S][768]
  constexpr int S = 4096;
  __shared__ __align__(16) _Float16 lds_k[32 * 64];     // [key][d]
  __shared__ __align__(16) _Float16 lds_v[64 * 32];     // [d][key]
  __shared__ __align__(16) _Float16 lds_p[4][16 * 32];  // per-wave P tile

  const int wave = threadIdx.x >> 5;
  const int lane = threadIdx.x & 31;
  const int head = blockIdx.y;
  const int qrow = blockIdx.x * 64 + wave * 16;
  const int nn = lane & 15;
  const int hs = lane >> 4;

  // q fragments for this wave's 16 rows (d = 0..31, 32..63)
  const v16h aq0 = load_a_frag(qbuf + head * 64, 768, qrow, 0);
  const v16h aq1 = load_a_frag(qbuf + head * 64, 768, qrow, 32);

  FragC o[4];
#pragma unroll
  for (int t = 0; t < 4; ++t)
#pragma unroll
    for (int r = 0; r < 8; ++r) o[t].f[r] = 0.0f;
  float m8[8], l8[8];
#pragma unroll
  for (int r = 0; r < 8; ++r) { m8[r] = -3.0e38f; l8[r] = 0.0f; }

  const int nkb_blk = blockIdx.x * 2 + 2;          // block-wide key blocks
  const int nkb_wv  = ((qrow + 16) + 31) >> 5;     // this wave's causal bound

  // staging index precompute
  const int kkey = threadIdx.x >> 2;               // 0..31
  const int kprt = (threadIdx.x & 3) * 16;         // 0,16,32,48
  const int vd   = threadIdx.x >> 1;               // 0..63
  const int vprt = (threadIdx.x & 1) * 16;         // 0,16

  for (int blk = 0; blk < nkb_blk; ++blk) {
    const int key0 = blk * 32;
    {  // K tile: 32 keys x 64 d (async global -> LDS)
      const _Float16* src = kbuf + (size_t)(key0 + kkey) * 768 + head * 64 + kprt;
      g2lds_b128(src,     &lds_k[kkey * 64 + kprt]);
      g2lds_b128(src + 8, &lds_k[kkey * 64 + kprt + 8]);
    }
    {  // V tile transposed: 64 d x 32 keys (async global -> LDS)
      const _Float16* src = vt + (size_t)(head * 64 + vd) * S + key0 + vprt;
      g2lds_b128(src,     &lds_v[vd * 32 + vprt]);
      g2lds_b128(src + 8, &lds_v[vd * 32 + vprt + 8]);
    }
    wait_async_all();
    __syncthreads();

    if (blk < nkb_wv) {  // wave-uniform
      // scores: two 16x16 tiles (keys key0..+15, key0+16..+31), K-dim = 64
      FragC s0, s1;
#pragma unroll
      for (int r = 0; r < 8; ++r) { s0.f[r] = 0.0f; s1.f[r] = 0.0f; }
      s0.v = wmma16(aq0, load_bT_frag(lds_k, 64, 0, 0), s0.v);
      s0.v = wmma16(aq1, load_bT_frag(lds_k, 64, 0, 32), s0.v);
      s1.v = wmma16(aq0, load_bT_frag(lds_k, 64, 16, 0), s1.v);
      s1.v = wmma16(aq1, load_bT_frag(lds_k, 64, 16, 32), s1.v);

      // online softmax per row; row M = r + 8*hs, N across 16-lane groups
#pragma unroll
      for (int r = 0; r < 8; ++r) {
        const int mg = qrow + r + hs * 8;
        float sc0 = s0.f[r] * 0.125f;  // 1/sqrt(64)
        float sc1 = s1.f[r] * 0.125f;
        if (key0 + nn > mg)      sc0 = -1.0e30f;
        if (key0 + 16 + nn > mg) sc1 = -1.0e30f;
        float mx = fmaxf(sc0, sc1);
        mx = fmaxf(mx, __shfl_xor(mx, 1));
        mx = fmaxf(mx, __shfl_xor(mx, 2));
        mx = fmaxf(mx, __shfl_xor(mx, 4));
        mx = fmaxf(mx, __shfl_xor(mx, 8));
        const float mnew = fmaxf(m8[r], mx);
        const float corr = __expf(m8[r] - mnew);
        const float p0 = __expf(sc0 - mnew);
        const float p1 = __expf(sc1 - mnew);
        float rs = p0 + p1;
        rs += __shfl_xor(rs, 1);
        rs += __shfl_xor(rs, 2);
        rs += __shfl_xor(rs, 4);
        rs += __shfl_xor(rs, 8);
        l8[r] = l8[r] * corr + rs;
        m8[r] = mnew;
        o[0].f[r] *= corr; o[1].f[r] *= corr;
        o[2].f[r] *= corr; o[3].f[r] *= corr;
        const int mrow = r + hs * 8;
        lds_p[wave][mrow * 32 + nn]      = (_Float16)p0;
        lds_p[wave][mrow * 32 + 16 + nn] = (_Float16)p1;
      }

      // O += P(16x32) @ V(32x64); per-wave LDS ops are in-order (RAW safe)
      const v16h ap = load_a_frag(&lds_p[wave][0], 32, 0, 0);
      o[0].v = wmma16(ap, load_bT_frag(lds_v, 32, 0, 0), o[0].v);
      o[1].v = wmma16(ap, load_bT_frag(lds_v, 32, 16, 0), o[1].v);
      o[2].v = wmma16(ap, load_bT_frag(lds_v, 32, 32, 0), o[2].v);
      o[3].v = wmma16(ap, load_bT_frag(lds_v, 32, 48, 0), o[3].v);
    }
    __syncthreads();
  }

  // normalize and store f16 attention output [S][H*64]
#pragma unroll
  for (int r = 0; r < 8; ++r) {
    const float inv = 1.0f / l8[r];
    _Float16* dst = attb + (size_t)(qrow + r + hs * 8) * 768 + head * 64 + nn;
    dst[0]  = (_Float16)(o[0].f[r] * inv);
    dst[16] = (_Float16)(o[1].f[r] * inv);
    dst[32] = (_Float16)(o[2].f[r] * inv);
    dst[48] = (_Float16)(o[3].f[r] * inv);
  }
}

// ---------------- launcher ----------------

extern "C" void kernel_launch(void* const* d_in, const int* in_sizes, int n_in,
                              void* d_out, int out_size, void* d_ws,
                              size_t ws_size, hipStream_t stream) {
  const float* Q  = (const float*)d_in[0];
  const float* K  = (const float*)d_in[1];
  const float* V  = (const float*)d_in[2];
  const float* Wq = (const float*)d_in[3];
  const float* Wk = (const float*)d_in[4];
  const float* Wv = (const float*)d_in[5];
  const float* Wo = (const float*)d_in[6];
  const float* bo = (const float*)d_in[7];
  float* out = (float*)d_out;

  constexpr int S = 4096, DM = 768;
  const size_t SD = (size_t)S * DM;   // 3,145,728 halves
  const size_t WW = (size_t)DM * DM;  //   589,824 halves

  _Float16* ws  = (_Float16*)d_ws;
  _Float16* Qh  = ws;        // f16 inputs
  _Float16* Kh  = Qh + SD;
  _Float16* Vh  = Kh + SD;
  _Float16* WqT = Vh + SD;   // transposed f16 weights
  _Float16* WkT = WqT + WW;
  _Float16* WvT = WkT + WW;
  _Float16* WoT = WvT + WW;
  _Float16* qp  = WoT + WW;  // q projection [S][768]
  _Float16* kp  = qp + SD;   // k projection [S][768]
  _Float16* vtp = Kh;        // alias: v^T projection [768][S]; Kh dead by then
  _Float16* att = Qh;        // alias: attention out [S][768]; Qh dead by then

  const int CT = 256;
  f32_to_f16_k<<<dim3((int)(SD / CT)), CT, 0, stream>>>(Q, Qh, (int)SD);
  f32_to_f16_k<<<dim3((int)(SD / CT)), CT, 0, stream>>>(K, Kh, (int)SD);
  f32_to_f16_k<<<dim3((int)(SD / CT)), CT, 0, stream>>>(V, Vh, (int)SD);
  wT_f32_to_f16_k<<<dim3((int)(WW / CT)), CT, 0, stream>>>(Wq, WqT);
  wT_f32_to_f16_k<<<dim3((int)(WW / CT)), CT, 0, stream>>>(Wk, WkT);
  wT_f32_to_f16_k<<<dim3((int)(WW / CT)), CT, 0, stream>>>(Wv, WvT);
  wT_f32_to_f16_k<<<dim3((int)(WW / CT)), CT, 0, stream>>>(Wo, WoT);

  dim3 gg(S / 128, DM / 64);  // (32, 12)
  gemm768_wmma<0><<<gg, 256, 0, stream>>>(Qh, WqT, qp, nullptr, nullptr, S);
  gemm768_wmma<0><<<gg, 256, 0, stream>>>(Kh, WkT, kp, nullptr, nullptr, S);
  gemm768_wmma<1><<<gg, 256, 0, stream>>>(Vh, WvT, vtp, nullptr, nullptr, S);

  attn_flash_wmma<<<dim3(S / 64, 12), 128, 0, stream>>>(qp, kp, vtp, att);

  gemm768_wmma<2><<<gg, 256, 0, stream>>>(att, WoT, nullptr, out, bo, S);
}
